// MinimalBP_39805756899418
// MI455X (gfx1250) — matrix operand order repaired and checked by reading, not compiled, not run on hardware
//
#include <hip/hip_runtime.h>
#include <hip/hip_bf16.h>

// ---------------------------------------------------------------------------
// Problem constants (B=8, T=2048, D=512)
// ---------------------------------------------------------------------------
#define BB   8
#define TT   2048
#define DD   512
#define MM   (BB * TT)          // 16384 rows
#define EPSC 1e-6f

typedef __attribute__((ext_vector_type(16))) __bf16 v16bf;
typedef __attribute__((ext_vector_type(8)))  __bf16 v8bf;
typedef __attribute__((ext_vector_type(8)))  float  v8f;
typedef __attribute__((ext_vector_type(4)))  unsigned v4u;

union V16U { v16bf v; v8bf h[2]; };

// Pack two f32 into two bf16 (round-to-nearest-even) in one 32-bit reg.
// Pure integer ops -> no 16-bit sub-register shuffling in codegen.
__device__ __forceinline__ unsigned pk_bf16x2(float a, float b) {
    unsigned ua = __float_as_uint(a);
    unsigned ub = __float_as_uint(b);
    ua += 0x7FFFu + ((ua >> 16) & 1u);
    ub += 0x7FFFu + ((ub >> 16) & 1u);
    return (ua >> 16) | (ub & 0xFFFF0000u);
}

__device__ __forceinline__ float gelu_exact(float x) {
    return 0.5f * x * (1.0f + erff(x * 0.70710678118654752f));
}

// ---------------------------------------------------------------------------
// Tiled bf16-WMMA GEMM:  C[m,n] = sum_k A[m,k] * Bw[n,k]   (A: MMxDD, Bw: DDxDD)
// Block tile 128(M) x 64(N), 8 waves = 4(M) x 2(N), wave tile 32x32
// EPI==0: store C -> Cout (hidden)
// EPI==1: h = GELU(C + b1[n]); partial logits[m] += h * W2[n] (atomic)
// ---------------------------------------------------------------------------
template<int EPI>
__launch_bounds__(256)
__global__ void gemm_bf16_wmma(const float* __restrict__ A,
                               const float* __restrict__ Bw,
                               float* __restrict__ Cout,
                               const float* __restrict__ bias1,
                               const float* __restrict__ W2,
                               float* __restrict__ logits) {
    // LDS tiles stored as packed bf16 pairs (uint): A 128x32, B 64x32
    __shared__ unsigned lds_a[128 * 16];   // 8 KB
    __shared__ unsigned lds_b[64 * 16];    // 4 KB

    const int tid  = threadIdx.x;
    const int lane = tid & 31;
    const int wave = tid >> 5;
    const int wm   = wave & 3;           // wave M index 0..3
    const int wn   = wave >> 2;          // wave N index 0..1
    const int row0 = blockIdx.x * 128;   // M block base
    const int col0 = blockIdx.y * 64;    // N block base
    const int r    = lane & 15;
    const int hs   = lane >> 4;          // lane half select

    v8f acc[2][2] = {};

    const int arow = tid >> 1;           // 0..127
    const int ah   = tid & 1;            // which 16-float half of the A row
    const int brow = tid >> 2;           // 0..63
    const int bq   = tid & 3;            // which 8-float quarter of the B row

    const float* srcA = A  + (size_t)(row0 + arow) * DD + ah * 16;
    const float* srcB = Bw + (size_t)(col0 + brow) * DD + bq * 8;

    for (int k0 = 0; k0 < DD; k0 += 32) {
        // prefetch next K tile into cache while we convert this one
        if (k0 + 32 < DD) {
            __builtin_prefetch(srcA + 32, 0, 3);
            __builtin_prefetch(srcB + 32, 0, 3);
        }
        // ---- load A tile 128x32 (f32 -> packed bf16x2) ----
        {
            float4 f0 = ((const float4*)srcA)[0];
            float4 f1 = ((const float4*)srcA)[1];
            float4 f2 = ((const float4*)srcA)[2];
            float4 f3 = ((const float4*)srcA)[3];
            v4u t0, t1;
            t0[0] = pk_bf16x2(f0.x, f0.y); t0[1] = pk_bf16x2(f0.z, f0.w);
            t0[2] = pk_bf16x2(f1.x, f1.y); t0[3] = pk_bf16x2(f1.z, f1.w);
            t1[0] = pk_bf16x2(f2.x, f2.y); t1[1] = pk_bf16x2(f2.z, f2.w);
            t1[2] = pk_bf16x2(f3.x, f3.y); t1[3] = pk_bf16x2(f3.z, f3.w);
            *(v4u*)(lds_a + arow * 16 + ah * 8)     = t0;
            *(v4u*)(lds_a + arow * 16 + ah * 8 + 4) = t1;
        }
        // ---- load B tile 64x32 (rows col0..col0+63 of Bw) ----
        {
            float4 f0 = ((const float4*)srcB)[0];
            float4 f1 = ((const float4*)srcB)[1];
            v4u t0;
            t0[0] = pk_bf16x2(f0.x, f0.y); t0[1] = pk_bf16x2(f0.z, f0.w);
            t0[2] = pk_bf16x2(f1.x, f1.y); t0[3] = pk_bf16x2(f1.z, f1.w);
            *(v4u*)(lds_b + brow * 16 + bq * 4) = t0;
        }
        srcA += 32;
        srcB += 32;
        __syncthreads();

        // ---- assemble fragments (contiguous k order: lane half h holds k 16h..16h+15) ----
        V16U afr[2];
        v16bf bfr[2];
#pragma unroll
        for (int mi = 0; mi < 2; ++mi) {
            // A 16-bit 16x32: lanes 0-15 (rows 0-15) K 0-15; lanes 16-31 K 16-31
            const unsigned* p = lds_a + (wm * 32 + mi * 16 + r) * 16 + hs * 8;
            afr[mi].h[0] = *(const v8bf*)(p);
            afr[mi].h[1] = *(const v8bf*)(p + 4);
        }
#pragma unroll
        for (int ni = 0; ni < 2; ++ni) {
            // B 16-bit 32x16: lanes 0-15 hold K 0-15 of col n; lanes 16-31 K 16-31
            bfr[ni] = *(const v16bf*)(lds_b + (wn * 32 + ni * 16 + r) * 16 + hs * 8);
        }
#pragma unroll
        for (int mi = 0; mi < 2; ++mi)
#pragma unroll
            for (int ni = 0; ni < 2; ++ni)
                acc[mi][ni] = __builtin_amdgcn_wmma_f32_16x16x32_bf16(
                    false, afr[mi].v, false, bfr[ni], (short)0, acc[mi][ni],
                    false, false);
        __syncthreads();
    }

    if (EPI == 0) {
        // write hidden (C layout: VGPR e -> row e (+8 for upper lane half), lane -> col)
#pragma unroll
        for (int mi = 0; mi < 2; ++mi)
#pragma unroll
            for (int ni = 0; ni < 2; ++ni) {
                const int n = col0 + wn * 32 + ni * 16 + r;
#pragma unroll
                for (int e = 0; e < 8; ++e) {
                    const int m = row0 + wm * 32 + mi * 16 + e + hs * 8;
                    Cout[(size_t)m * DD + n] = acc[mi][ni][e];
                }
            }
    } else {
        // fused bias + exact GELU + dot with W2, reduced into logits[m]
#pragma unroll
        for (int mi = 0; mi < 2; ++mi) {
            float s[8] = {0.f, 0.f, 0.f, 0.f, 0.f, 0.f, 0.f, 0.f};
#pragma unroll
            for (int ni = 0; ni < 2; ++ni) {
                const int n  = col0 + wn * 32 + ni * 16 + r;
                const float bv = bias1[n];
                const float wv = W2[n];
#pragma unroll
                for (int e = 0; e < 8; ++e)
                    s[e] += gelu_exact(acc[mi][ni][e] + bv) * wv;
            }
#pragma unroll
            for (int e = 0; e < 8; ++e) {
                float v = s[e];
                v += __shfl_xor(v, 1, 16);
                v += __shfl_xor(v, 2, 16);
                v += __shfl_xor(v, 4, 16);
                v += __shfl_xor(v, 8, 16);
                if (r == 0) {
                    const int m = row0 + wm * 32 + mi * 16 + e + hs * 8;
                    unsafeAtomicAdd(&logits[m], v);
                }
            }
        }
    }
}

// ---------------------------------------------------------------------------
// Zero-init small workspace array
// ---------------------------------------------------------------------------
__global__ void zero_f32(float* __restrict__ p, int n) {
    int i = blockIdx.x * blockDim.x + threadIdx.x;
    if (i < n) p[i] = 0.f;
}

// ---------------------------------------------------------------------------
// Per-row boundary decision + scan. One block per batch row (256 thr x 8 elems)
// ---------------------------------------------------------------------------
__launch_bounds__(256)
__global__ void boundary_kernel(const float* __restrict__ logits,
                                const float* __restrict__ u,
                                const float* __restrict__ b2p,
                                float* __restrict__ probs,
                                float* __restrict__ bnd,
                                int* __restrict__ seg_end,
                                int* __restrict__ nseg) {
    __shared__ int sums[256];
    __shared__ int pref[257];
    const int b   = blockIdx.x;
    const int tid = threadIdx.x;
    const float b2 = b2p[0];
    const int base = b * TT + tid * 8;

    int hv[8];
    int cnt = 0;
#pragma unroll
    for (int j = 0; j < 8; ++j) {
        const float l  = logits[base + j] + b2;
        const float pr = 1.f / (1.f + expf(-l));
        probs[base + j] = pr;
        const float p  = fminf(fmaxf(pr, EPSC), 1.f - EPSC);
        const float uu = fminf(fmaxf(u[base + j], EPSC), 1.f - EPSC);
        // soft = sigmoid(z/TEMP) > 0.5  <=>  z > 0
        const float z = logf(p) - log1pf(-p) + logf(uu) - log1pf(-uu);
        hv[j] = (z > 0.f) ? 1 : 0;
        cnt += hv[j];
    }
    sums[tid] = cnt;
    __syncthreads();
    if (tid == 0) {
        int run = 0;
        for (int i = 0; i < 256; ++i) { pref[i] = run; run += sums[i]; }
        pref[256] = run;
    }
    __syncthreads();
    const int total = pref[256];
    int run = pref[tid];
#pragma unroll
    for (int j = 0; j < 8; ++j) {
        const int t = tid * 8 + j;
        int h = hv[j];
        if (t == TT - 1 && total == 0) h = 1;      // guarantee >=1 boundary
        bnd[b * TT + t] = (float)h;
        if (h) { seg_end[b * TT + run] = t; ++run; }
    }
    if (tid == 0) nseg[b] = (total == 0) ? 1 : total;
}

// ---------------------------------------------------------------------------
// Segment mean pooling (segments are contiguous t-ranges => no atomics).
// One block per pooled row (b, s); 128 threads x float4 = 512 channels.
// Rows with s >= n_segments are written as zeros (covers required zero fill).
// ---------------------------------------------------------------------------
__launch_bounds__(128)
__global__ void pool_kernel(const float* __restrict__ hidden,
                            const int* __restrict__ seg_end,
                            const int* __restrict__ nseg,
                            float* __restrict__ pooled) {
    const int id = blockIdx.x;
    const int b  = id >> 11;           // /2048
    const int s  = id & (TT - 1);
    const int ns = nseg[b];

    float4 acc = make_float4(0.f, 0.f, 0.f, 0.f);
    if (s < ns) {
        const int start = s ? (seg_end[b * TT + s - 1] + 1) : 0;
        const int end   = seg_end[b * TT + s];
        const float4* src =
            (const float4*)(hidden + (size_t)(b * TT + start) * DD) + threadIdx.x;
        for (int t = start; t <= end; ++t) {
            float4 v = *src;
            acc.x += v.x; acc.y += v.y; acc.z += v.z; acc.w += v.w;
            src += DD / 4;
        }
        const float inv = 1.f / (float)(end - start + 1);
        acc.x *= inv; acc.y *= inv; acc.z *= inv; acc.w *= inv;
    }
    float4* dst = (float4*)(pooled + (size_t)(b * TT + s) * DD) + threadIdx.x;
    *dst = acc;
}

// ---------------------------------------------------------------------------
// Launch
// ---------------------------------------------------------------------------
extern "C" void kernel_launch(void* const* d_in, const int* in_sizes, int n_in,
                              void* d_out, int out_size, void* d_ws, size_t ws_size,
                              hipStream_t stream) {
    const float* x    = (const float*)d_in[0];   // [B,T,D]
    const float* u    = (const float*)d_in[1];   // [B,T]
    const float* W_up = (const float*)d_in[2];   // [D,D]
    const float* W1   = (const float*)d_in[3];   // [D,D]
    const float* b1   = (const float*)d_in[4];   // [D]
    const float* W2   = (const float*)d_in[5];   // [1,D]
    const float* b2   = (const float*)d_in[6];   // [1]

    float* out     = (float*)d_out;
    float* pooled  = out;                                  // B*T*D
    float* bnd     = out + (size_t)MM * DD;                // B*T
    float* probs   = bnd + MM;                             // B*T
    float* hidden  = probs + MM;                           // B*T*D

    float* logits  = (float*)d_ws;                         // MM floats
    int*   seg_end = (int*)d_ws + MM;                      // MM ints
    int*   nsegp   = (int*)d_ws + 2 * MM;                  // B ints

    // 0) zero logits accumulator
    zero_f32<<<(MM + 255) / 256, 256, 0, stream>>>(logits, MM);

    // 1) hidden = x @ W_up^T   (bf16 WMMA, f32 accumulate)
    gemm_bf16_wmma<0><<<dim3(MM / 128, DD / 64), 256, 0, stream>>>(
        x, W_up, hidden, nullptr, nullptr, nullptr);

    // 2) h = GELU(hidden @ W1^T + b1) fused into logits += h . W2
    gemm_bf16_wmma<1><<<dim3(MM / 128, DD / 64), 256, 0, stream>>>(
        hidden, W1, nullptr, b1, W2, logits);

    // 3) probs / boundaries / segment ranges
    boundary_kernel<<<BB, 256, 0, stream>>>(logits, u, b2, probs, bnd,
                                            seg_end, nsegp);

    // 4) segment mean pooling (+ zero fill of unused pooled rows)
    pool_kernel<<<BB * TT, 128, 0, stream>>>(hidden, seg_end, nsegp, pooled);
}
